// IPADecoder_40527311405663
// MI455X (gfx1250) — matrix operand order repaired and checked by reading, not compiled, not run on hardware
//
#include <hip/hip_runtime.h>
#include <hip/hip_bf16.h>

// ---------------------------------------------------------------------------
// IPA protein-structure decoder for MI455X (gfx1250, wave32, WMMA).
// All GEMMs run through V_WMMA_F32_16X16X32_F16 (f32 accumulate), activations
// converted fp32 -> f16 while staging to LDS. B tiles are stored transposed in
// LDS so both A and B fragment gathers are contiguous ds_load_b128s.
// ---------------------------------------------------------------------------

typedef __attribute__((ext_vector_type(16))) _Float16 v16h;
typedef __attribute__((ext_vector_type(8)))  float    v8f;

union HFrag { v16h v; _Float16 h[16]; };
union CFrag { v8f  v; float    f[8];  };

#define BB   2
#define NN   256
#define CS   256
#define CZ   128
#define HH   16
#define ROWS (BB*NN)          // 512
#define EROWS (BB*NN*NN)      // 131072
#define ET_HID_ 384
#define FEAT 3072             // H*(CH + 4*NV + CZ)

// ======================= generic WMMA GEMM ================================
// Y[M,N] = act(X[M,K] @ W[K,N] + bias) (+ res)   act: 0=none 1=relu
#define GT_M 128
#define GT_N 64
#define GT_K 32

__global__ __launch_bounds__(256)
void k_gemm(const float* __restrict__ X, const float* __restrict__ W,
            const float* __restrict__ bias, const float* res, float* Y,
            int M, int K, int N, int relu, int addres) {
  __shared__ _Float16 sA[GT_M][GT_K + 8];    // [row][k]   80B row stride
  __shared__ _Float16 sBt[GT_N][GT_K + 8];   // [col][k]   transposed B tile
  const int tid  = threadIdx.x;
  const int lane = tid & 31, wid = tid >> 5;
  const int wm = wid & 3, wn = wid >> 2;       // 4x2 wave grid -> 128x64 tile
  const int m0 = blockIdx.y * GT_M, n0 = blockIdx.x * GT_N;
  const int lr = lane & 15, lg = lane >> 4;

  v8f acc[2][2] = {};
  const int ksteps = (K + GT_K - 1) / GT_K;

  for (int ks = 0; ks < ksteps; ++ks) {
    const int k0 = ks * GT_K;
    const bool full = (m0 + GT_M <= M) && (n0 + GT_N <= N) && (k0 + GT_K <= K);
    __syncthreads();
    if (full) {
      // fast path: unconditional float4 loads (K, N multiples of 4; 16B align)
#pragma unroll
      for (int l = 0; l < 4; ++l) {
        int e4 = tid + l * 256;                // 1024 float4 = 128x32 floats
        int r = e4 >> 3, c4 = (e4 & 7) << 2;
        float4 xv = *reinterpret_cast<const float4*>(
            &X[(size_t)(m0 + r) * K + k0 + c4]);
        sA[r][c4 + 0] = (_Float16)xv.x; sA[r][c4 + 1] = (_Float16)xv.y;
        sA[r][c4 + 2] = (_Float16)xv.z; sA[r][c4 + 3] = (_Float16)xv.w;
      }
#pragma unroll
      for (int l = 0; l < 2; ++l) {
        int e4 = tid + l * 256;                // 512 float4 = 32x64 floats
        int kk = e4 >> 4, c4 = (e4 & 15) << 2;
        float4 wv = *reinterpret_cast<const float4*>(
            &W[(size_t)(k0 + kk) * N + n0 + c4]);
        sBt[c4 + 0][kk] = (_Float16)wv.x; sBt[c4 + 1][kk] = (_Float16)wv.y;
        sBt[c4 + 2][kk] = (_Float16)wv.z; sBt[c4 + 3][kk] = (_Float16)wv.w;
      }
      if (ks + 1 < ksteps) {                   // CDNA5 global_prefetch_b8
        __builtin_prefetch(
            &X[(size_t)(m0 + (tid >> 1)) * K + k0 + GT_K + ((tid & 1) << 4)], 0, 1);
        __builtin_prefetch(
            &W[(size_t)(k0 + GT_K + (tid >> 4)) * N + n0 + ((tid & 15) << 2)], 0, 1);
      }
    } else {
      // edge tiles: guarded scalar staging with zero padding
#pragma unroll
      for (int l = 0; l < 16; ++l) {
        int e = tid + l * 256;
        int r = e >> 5, kk = e & 31;
        float x = 0.f;
        if ((m0 + r) < M && (k0 + kk) < K) x = X[(size_t)(m0 + r) * K + k0 + kk];
        sA[r][kk] = (_Float16)x;
      }
#pragma unroll
      for (int l = 0; l < 8; ++l) {
        int e = tid + l * 256;
        int kk = e >> 6, c = e & 63;
        float x = 0.f;
        if ((k0 + kk) < K && (n0 + c) < N) x = W[(size_t)(k0 + kk) * N + n0 + c];
        sBt[c][kk] = (_Float16)x;
      }
    }
    __syncthreads();

    // gather fragments per ISA 7.12.2 layouts (contiguous -> ds_load_b128)
    HFrag af[2], bf[2];
#pragma unroll
    for (int am = 0; am < 2; ++am) {
      int row = wm * 32 + am * 16 + lr;
#pragma unroll
      for (int j = 0; j < 8; ++j) {
        af[am].h[j]     = sA[row][lg * 8 + j];        // K = 8*lg + j
        af[am].h[j + 8] = sA[row][16 + lg * 8 + j];   // K = 16 + 8*lg + j
      }
    }
#pragma unroll
    for (int bn = 0; bn < 2; ++bn) {
      int col = wn * 32 + bn * 16 + lr;
#pragma unroll
      for (int j = 0; j < 8; ++j) {
        bf[bn].h[j]     = sBt[col][lg * 8 + j];        // K = 8*lg + j
        bf[bn].h[j + 8] = sBt[col][16 + lg * 8 + j];   // K = 16 + 8*lg + j
      }
    }
#pragma unroll
    for (int am = 0; am < 2; ++am)
#pragma unroll
      for (int bn = 0; bn < 2; ++bn)
        acc[am][bn] = __builtin_amdgcn_wmma_f32_16x16x32_f16(
            false, af[am].v, false, bf[bn].v, (short)0, acc[am][bn], false, false);
  }

  // store D (lane: col = lr, VGPR r -> row r + 8*lg)
#pragma unroll
  for (int am = 0; am < 2; ++am)
#pragma unroll
    for (int bn = 0; bn < 2; ++bn) {
      int gc = n0 + wn * 32 + bn * 16 + lr;
      if (gc >= N) continue;
      CFrag cf; cf.v = acc[am][bn];
#pragma unroll
      for (int r = 0; r < 8; ++r) {
        int gm = m0 + wm * 32 + am * 16 + r + 8 * lg;
        if (gm >= M) continue;
        float vv = cf.f[r] + bias[gc];
        if (relu) vv = fmaxf(vv, 0.f);
        if (addres) vv += res[(size_t)gm * N + gc];
        Y[(size_t)gm * N + gc] = vv;
      }
    }
}

// ======================= LayerNorm (fused residual/mask) ===================
__global__ void k_layernorm(const float* X, const float* res, const float* mask,
                            const float* g, const float* b, float* Y,
                            int D, int premask, int postmask) {
  const int row = blockIdx.x, tid = threadIdx.x;
  __shared__ float red[256];
  float mv = mask ? mask[row] : 1.f;
  float v = X[(size_t)row * D + tid];
  if (premask) v *= mv;
  if (res) v += res[(size_t)row * D + tid];
  red[tid] = v; __syncthreads();
  for (int s = blockDim.x >> 1; s > 0; s >>= 1) {
    if (tid < s) red[tid] += red[tid + s];
    __syncthreads();
  }
  float mean = red[0] / (float)D; __syncthreads();
  float d = v - mean;
  red[tid] = d * d; __syncthreads();
  for (int s = blockDim.x >> 1; s > 0; s >>= 1) {
    if (tid < s) red[tid] += red[tid + s];
    __syncthreads();
  }
  float var = red[0] / (float)D;
  float y = d * rsqrtf(var + 1e-5f) * g[tid] + b[tid];
  if (postmask) y *= mv;
  Y[(size_t)row * D + tid] = y;
}

// ======================= small elementwise kernels =========================
__global__ void k_build_node_input(const float* ls, const float* t,
                                   const float* timeW, float* x) {
  int idx = blockIdx.x * blockDim.x + threadIdx.x;
  if (idx >= ROWS * 160) return;
  int row = idx / 160, c = idx % 160, b = row >> 8;
  float v;
  if (c < 64)       v = sinf(t[b] * timeW[c] * 6.28318530718f);
  else if (c < 128) v = cosf(t[b] * timeW[c - 64] * 6.28318530718f);
  else              v = ls[(size_t)row * 32 + (c - 128)];
  x[idx] = v;
}

__global__ void k_build_edge_input(const float* le, float* x) {
  size_t idx = (size_t)blockIdx.x * blockDim.x + threadIdx.x;
  if (idx >= (size_t)EROWS * 48) return;
  size_t row = idx / 48; int c = (int)(idx % 48);
  int ij = (int)(row % (NN * NN));
  int i = ij / NN, j = ij % NN;
  float v;
  if (c < 32) v = le[row * 32 + c];
  else {
    int f = (c - 32) & 7;
    float freq = expf((float)(2 * f) * (-9.21034037f / 16.f));
    float ang = (float)(i - j) * freq;
    v = ((c - 32) >= 8) ? sinf(ang) : cosf(ang);
  }
  x[idx] = v;
}

__global__ void k_quat_rot(const float* rig, float* rot, float* trans) {
  int r = blockIdx.x * blockDim.x + threadIdx.x;
  if (r >= ROWS) return;
  float w = rig[r*7+0], x = rig[r*7+1], y = rig[r*7+2], z = rig[r*7+3];
  float n = rsqrtf(w*w + x*x + y*y + z*z);
  w *= n; x *= n; y *= n; z *= n;
  float* R = rot + r * 9;
  R[0] = 1 - 2*(y*y + z*z); R[1] = 2*(x*y - w*z); R[2] = 2*(x*z + w*y);
  R[3] = 2*(x*y + w*z); R[4] = 1 - 2*(x*x + z*z); R[5] = 2*(y*z - w*x);
  R[6] = 2*(x*z - w*y); R[7] = 2*(y*z + w*x); R[8] = 1 - 2*(x*x + y*y);
  trans[r*3+0] = rig[r*7+4]; trans[r*3+1] = rig[r*7+5]; trans[r*3+2] = rig[r*7+6];
}

__global__ void k_point_glob(const float* pts, const float* rot, const float* trans,
                             float* out, int npts) {
  int idx = blockIdx.x * blockDim.x + threadIdx.x;
  int total = ROWS * HH * npts;
  if (idx >= total) return;
  int row = idx / (HH * npts);
  const float* p = pts + (size_t)idx * 3;
  const float* R = rot + row * 9;
  const float* t = trans + row * 3;
  float x = p[0], y = p[1], z = p[2];
  float* o = out + (size_t)idx * 3;
  o[0] = R[0]*x + R[1]*y + R[2]*z + t[0];
  o[1] = R[3]*x + R[4]*y + R[5]*z + t[1];
  o[2] = R[6]*x + R[7]*y + R[8]*z + t[2];
}

// one block per (b,h,i): logits over j + softmax
__global__ void k_attn_logits(const float* q, const float* k, const float* qpg,
                              const float* kpg, const float* bterm,
                              const float* head_w, const float* mask, float* attn) {
  int bid = blockIdx.x;
  int i = bid & 255, h = (bid >> 8) & 15, b = bid >> 12;
  int j = threadIdx.x;
  __shared__ float sq[16], sp[24], red[256];
  int rowi = b * NN + i;
  if (j < 16) sq[j] = q[(size_t)rowi * 256 + h * 16 + j];
  if (j < 24) sp[j] = qpg[((size_t)rowi * HH + h) * 24 + j];
  __syncthreads();
  int rowj = b * NN + j;
  float dot = 0.f;
#pragma unroll
  for (int c = 0; c < 16; ++c) dot += sq[c] * k[(size_t)rowj * 256 + h * 16 + c];
  float d2 = 0.f;
#pragma unroll
  for (int pd = 0; pd < 24; ++pd) {
    float df = sp[pd] - kpg[((size_t)rowj * HH + h) * 24 + pd];
    d2 += df * df;
  }
  float hw = log1pf(expf(head_w[h])) * 0.09622504486f;   // sqrt(1/(3*36))
  float mij = (mask[rowi] * mask[rowj] - 1.f) * 1e9f;
  float logit = dot * 0.14433756730f                     // sqrt(1/48)
              + 0.57735026919f * bterm[((size_t)rowi * NN + j) * HH + h]
              - 0.5f * hw * d2 + mij;
  red[j] = logit; __syncthreads();
  for (int s = 128; s > 0; s >>= 1) {
    if (j < s) red[j] = fmaxf(red[j], red[j + s]);
    __syncthreads();
  }
  float mx = red[0]; __syncthreads();
  float e = expf(logit - mx);
  red[j] = e; __syncthreads();
  for (int s = 128; s > 0; s >>= 1) {
    if (j < s) red[j] += red[j + s];
    __syncthreads();
  }
  attn[(size_t)bid * NN + j] = e / red[0];
}

// one block per (b,i): o, optl(+norm), opair(WMMA) -> feats[.,3072]
__global__ __launch_bounds__(256)
void k_attn_apply(const float* attn, const float* v, const float* vpg,
                  const float* z, const float* rot, const float* trans,
                  float* feats) {
  const int bi = blockIdx.x, b = bi >> 8, i = bi & 255;
  const int tid = threadIdx.x, lane = tid & 31, wid = tid >> 5;
  const int lr = lane & 15, lg = lane >> 4;
  __shared__ _Float16 sa[16][256];           // A: [h][j]
  __shared__ _Float16 szt[128][GT_K + 8];    // B tile transposed: [c][j]
  __shared__ float sopt[576];

  for (int e = tid; e < 16 * 256; e += 256) {
    int h = e >> 8, j = e & 255;
    sa[h][j] = (_Float16)attn[((((size_t)b * HH + h) * NN) + i) * NN + j];
  }
  __syncthreads();

  // opair: [16h x 128c] = sa(16x256) @ z[b,i](256x128), one 16-col tile / wave
  size_t zbase = (size_t)(b * NN + i) * NN * CZ;
  v8f acc = {};
  const int c0 = wid * 16;
  for (int kk = 0; kk < 8; ++kk) {
    __syncthreads();
#pragma unroll
    for (int l = 0; l < 4; ++l) {
      int e4 = tid + l * 256;                // 1024 float4 = 32x128 floats
      int j = e4 >> 5, c4 = (e4 & 31) << 2;
      float4 zv = *reinterpret_cast<const float4*>(
          &z[zbase + (size_t)(kk * 32 + j) * CZ + c4]);
      szt[c4 + 0][j] = (_Float16)zv.x; szt[c4 + 1][j] = (_Float16)zv.y;
      szt[c4 + 2][j] = (_Float16)zv.z; szt[c4 + 3][j] = (_Float16)zv.w;
    }
    __syncthreads();
    HFrag af, bf;
#pragma unroll
    for (int j = 0; j < 8; ++j) {
      af.h[j]     = sa[lr][kk * 32 + lg * 8 + j];
      af.h[j + 8] = sa[lr][kk * 32 + 16 + lg * 8 + j];
    }
#pragma unroll
    for (int j = 0; j < 8; ++j) {
      bf.h[j]     = szt[c0 + lr][lg * 8 + j];
      bf.h[j + 8] = szt[c0 + lr][16 + lg * 8 + j];
    }
    acc = __builtin_amdgcn_wmma_f32_16x16x32_f16(false, af.v, false, bf.v,
                                                 (short)0, acc, false, false);
  }
  size_t fb = (size_t)(b * NN + i) * FEAT;
  {
    CFrag cf; cf.v = acc;
#pragma unroll
    for (int r = 0; r < 8; ++r) {
      int h = r + 8 * lg;
      feats[fb + 1024 + (size_t)h * 128 + c0 + lr] = cf.f[r];
    }
  }
  __syncthreads();

  // o[h,c]: 256 outputs, one per thread
  {
    int h = tid >> 4, c = tid & 15;
    float s = 0.f;
    for (int j = 0; j < NN; ++j)
      s += (float)sa[h][j] * v[(size_t)(b * NN + j) * 256 + h * 16 + c];
    feats[fb + tid] = s;
  }
  // opt[h,p,d]: 576 outputs
  for (int idx = tid; idx < 576; idx += 256) {
    int h = idx / 36;
    float s = 0.f;
    for (int j = 0; j < NN; ++j)
      s += (float)sa[h][j] * vpg[(size_t)(b * NN + j) * 576 + idx];
    sopt[idx] = s;
  }
  __syncthreads();
  // optl = rot^T (opt - trans), norm
  if (tid < 192) {
    int h = tid / 12, p = tid % 12, rowi = b * NN + i;
    const float* R = rot + rowi * 9;
    float d0 = sopt[h*36 + p*3 + 0] - trans[rowi*3 + 0];
    float d1 = sopt[h*36 + p*3 + 1] - trans[rowi*3 + 1];
    float d2 = sopt[h*36 + p*3 + 2] - trans[rowi*3 + 2];
    float x = R[0]*d0 + R[3]*d1 + R[6]*d2;
    float y = R[1]*d0 + R[4]*d1 + R[7]*d2;
    float zz= R[2]*d0 + R[5]*d1 + R[8]*d2;
    feats[fb + 256 + h*36 + p*3 + 0] = x;
    feats[fb + 256 + h*36 + p*3 + 1] = y;
    feats[fb + 256 + h*36 + p*3 + 2] = zz;
    feats[fb + 832 + h*12 + p] = sqrtf(x*x + y*y + zz*zz + 1e-8f);
  }
}

__global__ void k_build_h(const float* e, const float* nb, float* h) {
  size_t idx = (size_t)blockIdx.x * blockDim.x + threadIdx.x;
  if (idx >= (size_t)EROWS * ET_HID_) return;
  size_t row = idx / ET_HID_; int c = (int)(idx % ET_HID_);
  int b = (int)(row / (NN * NN));
  int ij = (int)(row % (NN * NN));
  int i = ij / NN, j = ij % NN;
  float v;
  if (c < 128)      v = e[row * CZ + c];
  else if (c < 256) v = nb[(size_t)(b * NN + i) * 128 + (c - 128)];
  else              v = nb[(size_t)(b * NN + j) * 128 + (c - 256)];
  h[idx] = v;
}

__global__ void k_argmax(const float* logits, int* seq) {
  int r = blockIdx.x * blockDim.x + threadIdx.x;
  if (r >= ROWS) return;
  const float* L = logits + (size_t)r * 20;
  int best = 0; float bv = L[0];
  for (int c = 1; c < 20; ++c) if (L[c] > bv) { bv = L[c]; best = c; }
  seq[r] = best;
}

__global__ void k_build_tor_in(const float* nf, const float* emb,
                               const int* seq, float* x) {
  int idx = blockIdx.x * blockDim.x + threadIdx.x;
  if (idx >= ROWS * 276) return;
  int row = idx / 276, c = idx % 276;
  x[idx] = (c < 256) ? nf[(size_t)row * 256 + c]
                     : emb[seq[row] * 20 + (c - 256)];
}

__global__ void k_norm_tor(const float* u, float* un, float* chis_out) {
  int idx = blockIdx.x * blockDim.x + threadIdx.x;
  if (idx >= ROWS * 5) return;
  int row = idx / 5, k = idx % 5;
  float u0 = u[row*10 + k*2], u1 = u[row*10 + k*2 + 1];
  float n = fmaxf(sqrtf(u0*u0 + u1*u1), 1e-12f);
  u0 /= n; u1 /= n;
  un[row*10 + k*2] = u0; un[row*10 + k*2 + 1] = u1;
  if (k > 0) {
    chis_out[row*8 + (k-1)*2]     = u0;
    chis_out[row*8 + (k-1)*2 + 1] = u1;
  }
}

__device__ inline void mm3(const float* A, const float* B, float* C) {
#pragma unroll
  for (int i = 0; i < 3; ++i)
#pragma unroll
    for (int j = 0; j < 3; ++j)
      C[i*3+j] = A[i*3+0]*B[0*3+j] + A[i*3+1]*B[1*3+j] + A[i*3+2]*B[2*3+j];
}
__device__ inline void mv3(const float* A, const float* x, float* y) {
#pragma unroll
  for (int i = 0; i < 3; ++i)
    y[i] = A[i*3+0]*x[0] + A[i*3+1]*x[1] + A[i*3+2]*x[2];
}

__global__ void k_atom14(const float* rotg, const float* transg, const float* u,
                         const int* seq, const float* frame_rot,
                         const float* frame_trans, const int* group_idx,
                         const float* atom_mask, const float* lit,
                         float* out_atoms, float* out_mask) {
  int r = blockIdx.x * blockDim.x + threadIdx.x;
  if (r >= ROWS) return;
  int s = seq[r];
  float sn[8], cs[8];
  sn[0]=sn[1]=sn[2]=0.f; cs[0]=cs[1]=cs[2]=1.f;
  sn[3] = u[r*10+0]; cs[3] = u[r*10+1];
#pragma unroll
  for (int kc = 0; kc < 4; ++kc) { sn[4+kc] = u[r*10+2+2*kc]; cs[4+kc] = u[r*10+3+2*kc]; }
  float fr[8][9], ft[8][3];
#pragma unroll
  for (int f = 0; f < 8; ++f) {
    const float* dr = frame_rot + ((size_t)s * 8 + f) * 9;
    const float* dt = frame_trans + ((size_t)s * 8 + f) * 3;
    float c = cs[f], si = sn[f];
#pragma unroll
    for (int i = 0; i < 3; ++i) {
      fr[f][i*3+0] = dr[i*3+0];
      fr[f][i*3+1] = dr[i*3+1]*c + dr[i*3+2]*si;
      fr[f][i*3+2] = -dr[i*3+1]*si + dr[i*3+2]*c;
      ft[f][i] = dt[i];
    }
  }
  for (int f = 5; f < 8; ++f) {
    float tR[9], tT[3];
    mm3(fr[f-1], fr[f], tR);
    mv3(fr[f-1], ft[f], tT);
#pragma unroll
    for (int i = 0; i < 9; ++i) fr[f][i] = tR[i];
#pragma unroll
    for (int i = 0; i < 3; ++i) ft[f][i] = tT[i] + ft[f-1][i];
  }
  const float* Rg = rotg + r * 9;
  const float* tg = transg + r * 3;
  float gr[8][9], gt[8][3];
#pragma unroll
  for (int f = 0; f < 8; ++f) {
    mm3(Rg, fr[f], gr[f]);
    float tT[3]; mv3(Rg, ft[f], tT);
#pragma unroll
    for (int i = 0; i < 3; ++i) gt[f][i] = tT[i] + tg[i];
  }
  for (int a = 0; a < 14; ++a) {
    int g = group_idx[s * 14 + a];
    const float* lp = lit + ((size_t)s * 14 + a) * 3;
    float m = atom_mask[s * 14 + a];
    float p[3]; mv3(gr[g], lp, p);
#pragma unroll
    for (int d = 0; d < 3; ++d)
      out_atoms[((size_t)r * 14 + a) * 3 + d] = (p[d] + gt[g][d]) * m;
    if (out_mask) out_mask[r * 14 + a] = m;
  }
}

// ======================= host launcher =====================================
static inline int cdiv(int a, int b) { return (a + b - 1) / b; }

extern "C" void kernel_launch(void* const* d_in, const int* in_sizes, int n_in,
                              void* d_out, int out_size, void* d_ws, size_t ws_size,
                              hipStream_t stream) {
  (void)in_sizes; (void)n_in; (void)out_size; (void)ws_size;
  // Input flattening assumption: 6 tensors in setup_inputs dict order, then
  // params pytree leaves (alphabetical keys; each Linear -> {b,w}; LN -> {b,g};
  // layers list in order, layer dict {et,ipa,ln,nt} alphabetical), then consts
  // alphabetical.
  const float* ls    = (const float*)d_in[0];
  const float* le    = (const float*)d_in[1];
  const float* rig   = (const float*)d_in[2];
  const float* mask  = (const float*)d_in[3];
  const float* tvec  = (const float*)d_in[4];
  const int*   gtseq = (const int*)  d_in[5];
  auto P  = [&](int i) { return (const float*)d_in[i]; };
  auto PI = [&](int i) { return (const int*)  d_in[i]; };
  enum { EE1_B=6, EE1_W, EE2_B, EE2_W, EE3_B, EE3_W, EELN_B, EELN_G,
         EN1_B, EN1_W, EN2_B, EN2_W, LAYER0 = 18,
         NODELN_B = 166, NODELN_G, SEQ_EMBED, SEQH_B, SEQH_W, TIME_W,
         TOR_B, TOR_W, C_ATOM_MASK, C_FRAME_ROT, C_FRAME_TRANS,
         C_GROUP_IDX, C_LIT };

  // ---- workspace arena (all sizes multiples of 4 floats -> 16B alignment)
  float* ws = (float*)d_ws;
  size_t off = 0;
  auto A = [&](size_t n) { float* p = ws + off; off += n; return p; };
  float* rot   = A(ROWS * 9);
  float* trans = A(ROWS * 3);
  float* node  = A(ROWS * CS);
  float* tmp1  = A(ROWS * 512);
  float* tmp2  = A(ROWS * 512);
  float* xin   = A(ROWS * 160);
  float* qb    = A(ROWS * 256);
  float* kb    = A(ROWS * 256);
  float* vb    = A(ROWS * 256);
  float* qp    = A(ROWS * 384);
  float* kp    = A(ROWS * 384);
  float* vp    = A(ROWS * 576);
  float* qpg   = A(ROWS * 384);
  float* kpg   = A(ROWS * 384);
  float* vpg   = A(ROWS * 576);
  float* nb    = A(ROWS * 128);
  float* feats = A((size_t)ROWS * FEAT);
  float* updb  = A(ROWS * CS);
  float* torin = A(ROWS * 276);
  float* utmp  = A(ROWS * 10);
  float* unrm  = A(ROWS * 10);
  int*   seqbuf= (int*)A(ROWS);
  float* eb    = A((size_t)EROWS * CZ);
  float* bterm = A((size_t)EROWS * HH);
  float* attn  = A((size_t)BB * HH * NN * NN);
  float* bufA  = A((size_t)EROWS * ET_HID_);
  float* bufB  = A((size_t)EROWS * ET_HID_);

  float* OUT = (float*)d_out;
  float* out_atoms  = OUT;             // 512*14*3
  float* out_amask  = OUT + 21504;     // 512*14
  float* out_chis   = OUT + 28672;     // 512*4*2
  float* out_logits = OUT + 32768;     // 512*20
  float* out_atomsg = OUT + 43008;     // 512*14*3
  float* out_chisg  = OUT + 64512;     // 512*4*2

  auto gemm = [&](const float* X, const float* W, const float* bs,
                  const float* rs, float* Y, int M, int K, int N,
                  int relu, int addres) {
    dim3 g(cdiv(N, GT_N), cdiv(M, GT_M));
    k_gemm<<<g, 256, 0, stream>>>(X, W, bs, rs, Y, M, K, N, relu, addres);
  };

  // ---- node embedding
  k_build_node_input<<<cdiv(ROWS*160,256),256,0,stream>>>(ls, tvec, P(TIME_W), xin);
  gemm(xin, P(EN1_W), P(EN1_B), nullptr, tmp1, ROWS, 160, 512, 1, 0);
  gemm(tmp1, P(EN2_W), P(EN2_B), nullptr, tmp2, ROWS, 512, 256, 0, 0);
  k_layernorm<<<ROWS,256,0,stream>>>(tmp2, nullptr, mask, P(NODELN_G), P(NODELN_B),
                                     node, 256, 1, 0);
  // ---- edge embedding
  k_build_edge_input<<<cdiv(EROWS*48,256),256,0,stream>>>(le, bufB);
  gemm(bufB, P(EE1_W), P(EE1_B), nullptr, bufA, EROWS, 48, 256, 1, 0);
  gemm(bufA, P(EE2_W), P(EE2_B), nullptr, bufB, EROWS, 256, 256, 1, 0);
  gemm(bufB, P(EE3_W), P(EE3_B), nullptr, bufA, EROWS, 256, 128, 0, 0);
  k_layernorm<<<EROWS,128,0,stream>>>(bufA, nullptr, nullptr, P(EELN_G), P(EELN_B),
                                      eb, 128, 0, 0);
  // ---- rigid frames
  k_quat_rot<<<cdiv(ROWS,256),256,0,stream>>>(rig, rot, trans);

  // ---- 4 IPA layers
  for (int l = 0; l < 4; ++l) {
    int LB = LAYER0 + 37 * l;
    const float *et_final_b=P(LB+0), *et_final_w=P(LB+1);
    const float *et_init_b =P(LB+2), *et_init_w =P(LB+3);
    const float *et_ln_b   =P(LB+4), *et_ln_g   =P(LB+5);
    const float *et_t1_b   =P(LB+6), *et_t1_w   =P(LB+7);
    const float *et_t2_b   =P(LB+8), *et_t2_w   =P(LB+9);
    const float *bz_b=P(LB+10), *bz_w=P(LB+11), *head_w=P(LB+12);
    const float *k_b=P(LB+13), *k_w=P(LB+14);
    const float *kp_b=P(LB+15), *kp_w=P(LB+16);
    const float *out_b=P(LB+17), *out_w=P(LB+18);
    const float *q_b=P(LB+19), *q_w=P(LB+20);
    const float *qp_b=P(LB+21), *qp_w=P(LB+22);
    const float *v_b=P(LB+23), *v_w=P(LB+24);
    const float *vp_b=P(LB+25), *vp_w=P(LB+26);
    const float *ln_b=P(LB+27), *ln_g=P(LB+28);
    const float *l1_b=P(LB+29), *l1_w=P(LB+30);
    const float *l2_b=P(LB+31), *l2_w=P(LB+32);
    const float *l3_b=P(LB+33), *l3_w=P(LB+34);
    const float *ntln_b=P(LB+35), *ntln_g=P(LB+36);

    gemm(node, q_w,  q_b,  nullptr, qb, ROWS, 256, 256, 0, 0);
    gemm(node, k_w,  k_b,  nullptr, kb, ROWS, 256, 256, 0, 0);
    gemm(node, v_w,  v_b,  nullptr, vb, ROWS, 256, 256, 0, 0);
    gemm(node, qp_w, qp_b, nullptr, qp, ROWS, 256, 384, 0, 0);
    gemm(node, kp_w, kp_b, nullptr, kp, ROWS, 256, 384, 0, 0);
    gemm(node, vp_w, vp_b, nullptr, vp, ROWS, 256, 576, 0, 0);
    gemm(eb, bz_w, bz_b, nullptr, bterm, EROWS, 128, 16, 0, 0);

    k_point_glob<<<cdiv(ROWS*HH*8,256),256,0,stream>>>(qp, rot, trans, qpg, 8);
    k_point_glob<<<cdiv(ROWS*HH*8,256),256,0,stream>>>(kp, rot, trans, kpg, 8);
    k_point_glob<<<cdiv(ROWS*HH*12,256),256,0,stream>>>(vp, rot, trans, vpg, 12);

    k_attn_logits<<<BB*HH*NN,256,0,stream>>>(qb, kb, qpg, kpg, bterm, head_w,
                                             mask, attn);
    k_attn_apply<<<BB*NN,256,0,stream>>>(attn, vb, vpg, eb, rot, trans, feats);
    gemm(feats, out_w, out_b, nullptr, updb, ROWS, FEAT, 256, 0, 0);
    k_layernorm<<<ROWS,256,0,stream>>>(updb, node, mask, ln_g, ln_b,
                                       node, 256, 1, 0);
    // node transition
    gemm(node, l1_w, l1_b, nullptr, tmp1, ROWS, 256, 256, 1, 0);
    gemm(tmp1, l2_w, l2_b, nullptr, tmp2, ROWS, 256, 256, 1, 0);
    gemm(tmp2, l3_w, l3_b, node,    tmp1, ROWS, 256, 256, 0, 1);
    k_layernorm<<<ROWS,256,0,stream>>>(tmp1, nullptr, mask, ntln_g, ntln_b,
                                       node, 256, 0, 1);
    // edge transition
    gemm(node, et_init_w, et_init_b, nullptr, nb, ROWS, 256, 128, 0, 0);
    k_build_h<<<cdiv(EROWS*ET_HID_,256),256,0,stream>>>(eb, nb, bufA);
    gemm(bufA, et_t1_w, et_t1_b, nullptr, bufB, EROWS, ET_HID_, ET_HID_, 1, 0);
    gemm(bufB, et_t2_w, et_t2_b, bufA,    bufA, EROWS, ET_HID_, ET_HID_, 1, 1);
    gemm(bufA, et_final_w, et_final_b, nullptr, bufB, EROWS, ET_HID_, 128, 0, 0);
    k_layernorm<<<EROWS,128,0,stream>>>(bufB, nullptr, nullptr, et_ln_g, et_ln_b,
                                        eb, 128, 0, 0);
  }

  // ---- heads
  gemm(node, P(SEQH_W), P(SEQH_B), nullptr, out_logits, ROWS, 256, 20, 0, 0);
  k_argmax<<<cdiv(ROWS,256),256,0,stream>>>(out_logits, seqbuf);

  // predicted-seq torsions + atoms
  k_build_tor_in<<<cdiv(ROWS*276,256),256,0,stream>>>(node, P(SEQ_EMBED), seqbuf, torin);
  gemm(torin, P(TOR_W), P(TOR_B), nullptr, utmp, ROWS, 276, 10, 0, 0);
  k_norm_tor<<<cdiv(ROWS*5,256),256,0,stream>>>(utmp, unrm, out_chis);
  k_atom14<<<cdiv(ROWS,64),64,0,stream>>>(rot, trans, unrm, seqbuf,
      P(C_FRAME_ROT), P(C_FRAME_TRANS), PI(C_GROUP_IDX), P(C_ATOM_MASK),
      P(C_LIT), out_atoms, out_amask);

  // gt-seq torsions + atoms
  k_build_tor_in<<<cdiv(ROWS*276,256),256,0,stream>>>(node, P(SEQ_EMBED), gtseq, torin);
  gemm(torin, P(TOR_W), P(TOR_B), nullptr, utmp, ROWS, 276, 10, 0, 0);
  k_norm_tor<<<cdiv(ROWS*5,256),256,0,stream>>>(utmp, unrm, out_chisg);
  k_atom14<<<cdiv(ROWS,64),64,0,stream>>>(rot, trans, unrm, gtseq,
      P(C_FRAME_ROT), P(C_FRAME_TRANS), PI(C_GROUP_IDX), P(C_ATOM_MASK),
      P(C_LIT), out_atomsg, nullptr);
}
